// EnhancedCausalTransformer_10350871183767
// MI455X (gfx1250) — matrix operand order repaired and checked by reading, not compiled, not run on hardware
//
#include <hip/hip_runtime.h>
#include <hip/hip_bf16.h>
#include <math.h>

typedef __attribute__((ext_vector_type(16))) _Float16 v16h;
typedef __attribute__((ext_vector_type(8)))  float    v8f;
typedef __attribute__((ext_vector_type(8)))  _Float16 h8;

// ---------------- output layout (flat, reference tuple order) ----------------
#define OUT_POL    0
#define OUT_VAL    8
#define OUT_CF     9
#define OUT_CM     17
#define OUT_CP     4113
#define OUT_INSTR  185553
#define OUT_PARSE  186577
#define OUT_ALIGN  186581
#define OUT_GL     186589
#define OUT_TL     186590

// ---------------- workspace layout (bytes, all 256B aligned) -----------------
constexpr size_t WS_WM1   = 0;                    // W_m1 f16  [1024][2048]  4 MB
constexpr size_t WS_WIH   = WS_WM1   + 4194304;   // W_ih f16  [4096][1024]  8 MB
constexpr size_t WS_WHH   = WS_WIH   + 8388608;   // W_hh f16  [4096][1024]  8 MB
constexpr size_t WS_OBJ   = WS_WHH   + 8388608;   // obj_emb f16 [64][1024]
constexpr size_t WS_X16   = WS_OBJ   + 131072;    // gathered tokens f16 [128][1024]
constexpr size_t WS_U     = WS_X16   + 262144;    // U f32 [64][1024]
constexpr size_t WS_V     = WS_U     + 262144;    // V f32 [64][1024]
constexpr size_t WS_XG    = WS_V     + 262144;    // xg f32 [128][4096]  2 MB
constexpr size_t WS_BCOMB = WS_XG    + 2097152;   // b_ih+b_hh [4096]
constexpr size_t WS_PRES  = WS_BCOMB + 16384;     // present [15][64] i32
constexpr size_t WS_CHG   = WS_PRES  + 4096;      // changed [15][64] i32
constexpr size_t WS_CM    = WS_CHG   + 4096;      // causal_mat f32 [64][64]
constexpr size_t WS_INSTR = WS_CM    + 16384;     // instr [1024]
constexpr size_t WS_SREP  = WS_INSTR + 4096;      // state_repr [1024]
constexpr size_t WS_CREP  = WS_SREP  + 4096;      // causal_repr [1024]
constexpr size_t WS_COMB3 = WS_CREP  + 4096;      // combined [3072]
constexpr size_t WS_COMB  = WS_COMB3 + 16384;     // comb [8][2048]
constexpr size_t WS_HPOL  = WS_COMB  + 65536;     // hidden pol [1024]
constexpr size_t WS_HVAL  = WS_HPOL  + 4096;
constexpr size_t WS_HCF   = WS_HVAL  + 4096;
constexpr size_t WS_HPAR  = WS_HCF   + 4096;
constexpr size_t WS_HAL   = WS_HPAR  + 4096;      // hidden align [8][1024]
constexpr size_t WS_SCAL  = WS_HAL   + 32768;     // [0]=loss_cm, [1]=grounding
constexpr size_t WS_HBUF  = WS_SCAL  + 256;       // double-buffered h [2][1024] f32
constexpr size_t WS_CNT   = WS_HBUF  + 8192;      // arrival counter

#define LSTM_NWG 16

// =============================================================================
// f16 WMMA GEMM:  C[M,N] = A[M,K] * W[N,K]^T (+bias[n]) (opt relu)
// One wave per 16x16 C tile.  K % 32 == 0.  A,W row-major f16, 16B aligned.
// Operand layouts per CDNA5 ISA 7.12.2:
//   A (16x32 f16): lane(l&15)=row m, half=l>>4; VGPR v holds k = 2v+(v>=4?8:0)+8*half
//   B (32x16 f16): lane(l&15)=col n, half=l>>4; VGPR v holds k = 2v+(v>=4?8:0)+8*half
//   C/D (16x16 f32): VGPR r -> row r + 8*half, col = lane&15
// =============================================================================
__global__ void gemm_f16_wmma(const _Float16* __restrict__ A, int lda,
                              const _Float16* __restrict__ W, int ldw,
                              float* __restrict__ C, int ldc,
                              int M, int N, int K,
                              const float* __restrict__ bias, int relu)
{
    const int lane = threadIdx.x & 31;
    const int wid  = (blockIdx.x * blockDim.x + threadIdx.x) >> 5;
    const int tn = N >> 4, tm = M >> 4;
    if (wid >= tm * tn) return;
    const int tile_m = wid / tn, tile_n = wid % tn;
    const int half = lane >> 4;
    const int mr = (tile_m << 4) + (lane & 15);
    const int nr = (tile_n << 4) + (lane & 15);
    const _Float16* arow = A + (size_t)mr * lda;
    const _Float16* wrow = W + (size_t)nr * ldw;

    v8f acc = {};
    for (int k0 = 0; k0 < K; k0 += 32) {
        const h8* pa = (const h8*)(arow + k0 + 8 * half);
        h8 a0 = pa[0];
        h8 a1 = pa[2];                         // +16 halves
        const h8* pb = (const h8*)(wrow + k0 + 8 * half);
        h8 b0 = pb[0], b1 = pb[2];
        v16h av, bv;
#pragma unroll
        for (int i = 0; i < 8; i++) {
            av[i] = a0[i]; av[8 + i] = a1[i];
            bv[i] = b0[i]; bv[8 + i] = b1[i];
        }
        acc = __builtin_amdgcn_wmma_f32_16x16x32_f16(false, av, false, bv,
                                                     (short)0, acc, false, false);
    }
    const int col   = (tile_n << 4) + (lane & 15);
    const int rbase = (tile_m << 4) + 8 * half;
#pragma unroll
    for (int r = 0; r < 8; r++) {
        float v = acc[r];
        if (bias) v += bias[col];
        if (relu) v = fmaxf(v, 0.0f);
        C[(size_t)(rbase + r) * ldc + col] = v;
    }
}

// =============================================================================
// batched GEMV: Y[m,n] = act( sum_k X[m,k]*W[n,k] + bias[n] ). One wave/output.
// K % 128 == 0 at all call sites.
// =============================================================================
__global__ void gemv_mb(const float* __restrict__ X, int ldx,
                        const float* __restrict__ W, int ldw,
                        const float* __restrict__ bias,
                        float* __restrict__ Y, int ldy,
                        int M, int N, int K, int relu)
{
    const int lane = threadIdx.x & 31;
    const int wid  = (blockIdx.x * blockDim.x + threadIdx.x) >> 5;
    if (wid >= M * N) return;
    const int m = wid / N, n = wid % N;
    const float4* xr = (const float4*)(X + (size_t)m * ldx);
    const float4* wr = (const float4*)(W + (size_t)n * ldw);
    const int k4n = K >> 2;
    float acc = 0.f;
    for (int k4 = lane; k4 < k4n; k4 += 32) {
        __builtin_prefetch(&wr[k4 + 64], 0, 0);
        float4 x = xr[k4]; float4 w = wr[k4];
        acc += x.x * w.x + x.y * w.y + x.z * w.z + x.w * w.w;
    }
#pragma unroll
    for (int off = 16; off > 0; off >>= 1) acc += __shfl_xor(acc, off, 32);
    if (lane == 0) {
        float v = acc + (bias ? bias[n] : 0.f);
        if (relu) v = fmaxf(v, 0.f);
        Y[(size_t)m * ldy + n] = v;
    }
}

// =============================================================================
// small utility kernels
// =============================================================================
__global__ void cvt_f32_f16(const float* __restrict__ in, _Float16* __restrict__ out, int n)
{
    int i = blockIdx.x * blockDim.x + threadIdx.x;
    if (i < n) out[i] = (_Float16)in[i];
}

__global__ void gather_tokens(const int* __restrict__ tok, const float* __restrict__ emb,
                              _Float16* __restrict__ x16)
{
    int i = blockIdx.x * blockDim.x + threadIdx.x;    // 128*1024
    int t = i >> 10, d = i & 1023;
    x16[i] = (_Float16)emb[(size_t)tok[t] * 1024 + d];
}

__global__ void vec_add(const float* __restrict__ a, const float* __restrict__ b,
                        float* __restrict__ o, int n)
{
    int i = blockIdx.x * blockDim.x + threadIdx.x;
    if (i < n) o[i] = a[i] + b[i];
}

__global__ void zero_int(int* __restrict__ p) { p[0] = 0; }

// mech head: per pair p, out3 = relu(U[o1]+V[o2]+b_m1) @ W_m2^T + b_m2,
// written tiled 15x into causal_predictions.
__global__ void mech_head(const float* __restrict__ U, const float* __restrict__ V,
                          const float* __restrict__ b_m1,
                          const float* __restrict__ W_m2, const float* __restrict__ b_m2,
                          float* __restrict__ out_cp)
{
    __shared__ float red[8][3];
    __shared__ float fin[3];
    const int p   = blockIdx.x;                 // 0..4031
    const int i   = p / 63;
    const int jj  = p % 63;
    const int j   = jj + (jj >= i ? 1 : 0);
    const int tid = threadIdx.x;                // 256
    const float* u = U + (size_t)i * 1024;
    const float* v = V + (size_t)j * 1024;
    float a0 = 0.f, a1 = 0.f, a2 = 0.f;
    for (int k = tid; k < 1024; k += 256) {
        float hv = fmaxf(u[k] + v[k] + b_m1[k], 0.f);
        a0 += hv * W_m2[k];
        a1 += hv * W_m2[1024 + k];
        a2 += hv * W_m2[2048 + k];
    }
#pragma unroll
    for (int off = 16; off > 0; off >>= 1) {
        a0 += __shfl_xor(a0, off, 32);
        a1 += __shfl_xor(a1, off, 32);
        a2 += __shfl_xor(a2, off, 32);
    }
    if ((tid & 31) == 0) { red[tid >> 5][0] = a0; red[tid >> 5][1] = a1; red[tid >> 5][2] = a2; }
    __syncthreads();
    if (tid < 3) {
        float s = b_m2[tid];
        for (int w = 0; w < 8; w++) s += red[w][tid];
        fin[tid] = s;
    }
    __syncthreads();
    if (tid < 45) {
        int r = tid / 3, c = tid % 3;
        out_cp[(size_t)(r * 4032 + p) * 3 + c] = fin[c];
    }
}

// present/changed per (t,obj) — reproduces reference semantics exactly
// (changed indexes the 0/1 diff values, so only obj 0 and 1 can be "changed").
__global__ void history_scan(const int* __restrict__ hist,
                             int* __restrict__ pres, int* __restrict__ chg)
{
    __shared__ int p_s[64], c_s[64];
    const int t = blockIdx.x;                   // 0..14
    const int tid = threadIdx.x;                // 256
    if (tid < 64) { p_s[tid] = 0; c_s[tid] = 0; }
    __syncthreads();
    const int* prev = hist + (size_t)t * 4096;
    const int* curr = prev + 4096;
    for (int px = tid; px < 4096; px += 256) {
        int pv = prev[px] & 63;
        p_s[pv] = 1;
        int d = (curr[px] != prev[px]) ? 1 : 0;
        c_s[d] = 1;
    }
    __syncthreads();
    if (tid < 64) { pres[t * 64 + tid] = p_s[tid]; chg[t * 64 + tid] = c_s[tid]; }
}

// delta -> sigmoid -> causal_mat, plus matrix-loss partial (mse vs eye + 0.1*L1)
__global__ void __launch_bounds__(1024) causal_mat_k(
    const float* __restrict__ causal_matrix,
    const int* __restrict__ pres, const int* __restrict__ chg,
    float* __restrict__ cm_ws, float* __restrict__ out_cm, float* __restrict__ scal)
{
    __shared__ float r1[32], r2[32];
    const int tid = threadIdx.x;                // 1024
    float le = 0.f, la = 0.f;
#pragma unroll
    for (int e = 0; e < 4; e++) {
        int idx = tid + e * 1024;
        int i = idx >> 6, j = idx & 63;
        int s = 0;
        for (int t = 0; t < 15; t++) s += (pres[t * 64 + i] & chg[t * 64 + j]);
        float delta = (i != j) ? 0.01f * (float)s : 0.f;
        float x = causal_matrix[idx] + delta;
        float cmv = 1.f / (1.f + __expf(-x));
        cm_ws[idx] = cmv; out_cm[idx] = cmv;
        float d = cmv - ((i == j) ? 1.f : 0.f);
        le += d * d; la += fabsf(cmv);
    }
#pragma unroll
    for (int off = 16; off > 0; off >>= 1) { le += __shfl_xor(le, off, 32); la += __shfl_xor(la, off, 32); }
    if ((tid & 31) == 0) { r1[tid >> 5] = le; r2[tid >> 5] = la; }
    __syncthreads();
    if (tid == 0) {
        float se = 0.f, sa = 0.f;
        for (int w = 0; w < 32; w++) { se += r1[w]; sa += r2[w]; }
        scal[0] = se / 4096.f + 0.1f * sa / 4096.f;
    }
}

// =============================================================================
// Multi-workgroup sequential LSTM (128 steps), 16 co-resident workgroups.
// WG w owns cells [64w,64w+64): 256 gate rows -> 512KB f16 weights/step/WGP.
// Thread t: gate-row (t>>2), K-chunk (t&3) of 256; 4-lane shfl reduce -> LDS.
// Cross-WG handoff: double-buffered global h + device-scope arrival counter
// (release fetch_add / acquire spin, agent-scope loads bypass stale WGP$).
// =============================================================================
__global__ void __launch_bounds__(1024) lstm_seq_mw(
    const float* __restrict__ xg, const _Float16* __restrict__ Whh,
    float* __restrict__ hbuf,     // [2][1024]
    int* __restrict__ cnt,
    float* __restrict__ instr_ws, float* __restrict__ out_instr)
{
    __shared__ float hsh[1024];
    __shared__ float g_sh[256];
    const int tid    = threadIdx.x;           // 1024
    const int wg     = blockIdx.x;            // 0..15
    const int wgbase = wg * 64;
    const int row_idx = tid >> 2;             // 0..255  (gate*64 + cell)
    const int kchunk  = tid & 3;
    const int gate = row_idx >> 6;
    const int cell = row_idx & 63;
    const int grow = gate * 1024 + wgbase + cell;
    const h8* wp = (const h8*)(Whh + (size_t)grow * 1024 + kchunk * 256);
    const int hbase = kchunk * 256;

    hsh[tid] = 0.f;
    float cst = 0.f, hsum = 0.f;
    __syncthreads();

    for (int t = 0; t < 128; t++) {
        // ---- partial dot over this thread's 256-wide K chunk ----
        float acc = 0.f;
#pragma unroll 4
        for (int kk = 0; kk < 32; kk++) {
            h8 w = wp[kk];
#pragma unroll
            for (int u = 0; u < 8; u++)
                acc += (float)w[u] * hsh[hbase + kk * 8 + u];
        }
        acc += __shfl_xor(acc, 1, 32);
        acc += __shfl_xor(acc, 2, 32);
        if (kchunk == 0) g_sh[row_idx] = acc;
        __syncthreads();

        // ---- cell update (threads 0..63 own persistent c-state) ----
        float* hout = hbuf + (t & 1) * 1024;
        if (tid < 64) {
            const float* xr = xg + (size_t)t * 4096 + wgbase + tid;
            float g0 = xr[0]    + g_sh[tid];
            float g1 = xr[1024] + g_sh[64 + tid];
            float g2 = xr[2048] + g_sh[128 + tid];
            float g3 = xr[3072] + g_sh[192 + tid];
            float i_ = 1.f / (1.f + __expf(-g0));
            float f_ = 1.f / (1.f + __expf(-g1));
            float gg = tanhf(g2);
            float o_ = 1.f / (1.f + __expf(-g3));
            cst = f_ * cst + i_ * gg;
            float hn = o_ * tanhf(cst);
            hsum += hn;
            __hip_atomic_store(&hout[wgbase + tid], hn, __ATOMIC_RELAXED,
                               __HIP_MEMORY_SCOPE_AGENT);
        }
        __threadfence();                       // own stores visible device-wide
        __syncthreads();
        if (tid == 0) {
            __hip_atomic_fetch_add(cnt, 1, __ATOMIC_RELEASE, __HIP_MEMORY_SCOPE_AGENT);
            const int target = LSTM_NWG * (t + 1);
            while (__hip_atomic_load(cnt, __ATOMIC_ACQUIRE, __HIP_MEMORY_SCOPE_AGENT) < target)
                __builtin_amdgcn_s_sleep(1);
        }
        __syncthreads();
        // ---- reload full h for next step (agent scope, bypass WGP$) ----
        hsh[tid] = __hip_atomic_load(&hout[tid], __ATOMIC_RELAXED,
                                     __HIP_MEMORY_SCOPE_AGENT);
        __syncthreads();
    }
    if (tid < 64) {
        float v = hsum * (1.f / 128.f);
        instr_ws[wgbase + tid]  = v;
        out_instr[wgbase + tid] = v;
    }
}

__global__ void build_comb(const float* __restrict__ instr, const int* __restrict__ actions,
                           const float* __restrict__ act_emb, float* __restrict__ comb)
{
    int i = blockIdx.x * blockDim.x + threadIdx.x;   // 8*2048
    int a = i >> 11, k = i & 2047;
    comb[i] = (k < 1024) ? instr[k] : act_emb[(size_t)actions[a] * 1024 + (k - 1024)];
}

__global__ void build_combined(const float* __restrict__ s, const float* __restrict__ cr,
                               const float* __restrict__ ins, float* __restrict__ comb3)
{
    int i = blockIdx.x * blockDim.x + threadIdx.x;   // 3072
    if (i < 1024) comb3[i] = s[i];
    else if (i < 2048) comb3[i] = cr[i - 1024];
    else comb3[i] = ins[i - 2048];
}

__global__ void __launch_bounds__(1024) grounding_k(
    const float* __restrict__ instr, const int* __restrict__ actions,
    const float* __restrict__ act_emb, float* __restrict__ out_gl, float* __restrict__ scal)
{
    __shared__ float r1[32], r2[32];
    const int tid = threadIdx.x;                // 1024
    float si = instr[tid];
    float sa = 0.f;
    for (int a = 0; a < 8; a++) sa += act_emb[(size_t)actions[a] * 1024 + tid];
#pragma unroll
    for (int off = 16; off > 0; off >>= 1) { si += __shfl_xor(si, off, 32); sa += __shfl_xor(sa, off, 32); }
    if ((tid & 31) == 0) { r1[tid >> 5] = si; r2[tid >> 5] = sa; }
    __syncthreads();
    if (tid == 0) {
        float S1 = 0.f, S2 = 0.f;
        for (int w = 0; w < 32; w++) { S1 += r1[w]; S2 += r2[w]; }
        float gl = (S1 / 1024.f - S2 / 8192.f);
        gl *= gl;
        out_gl[0] = gl; scal[1] = gl;
    }
}

__global__ void final_loss(const float* __restrict__ scal, float* __restrict__ out_tl)
{
    out_tl[0] = scal[0] + scal[1];
}

// =============================================================================
extern "C" void kernel_launch(void* const* d_in, const int* in_sizes, int n_in,
                              void* d_out, int out_size, void* d_ws, size_t ws_size,
                              hipStream_t stream)
{
    (void)in_sizes; (void)n_in; (void)out_size; (void)ws_size;
    // ---- inputs (setup_inputs order) ----
    const float* state    = (const float*)d_in[0];
    const int*   tokens   = (const int*)  d_in[1];
    const int*   actions  = (const int*)  d_in[2];
    const int*   hist     = (const int*)  d_in[3];
    const float* W_enc    = (const float*)d_in[4];
    const float* b_enc    = (const float*)d_in[5];
    const float* causal_m = (const float*)d_in[6];
    const float* obj_emb  = (const float*)d_in[7];
    const float* W_m1     = (const float*)d_in[8];
    const float* b_m1     = (const float*)d_in[9];
    const float* W_m2     = (const float*)d_in[10];
    const float* b_m2     = (const float*)d_in[11];
    const float* W_rand   = (const float*)d_in[12];
    const float* word_emb = (const float*)d_in[13];
    const float* act_emb  = (const float*)d_in[14];
    const float* W_ih     = (const float*)d_in[15];
    const float* W_hh     = (const float*)d_in[16];
    const float* b_ih     = (const float*)d_in[17];
    const float* b_hh     = (const float*)d_in[18];
    const float* W_a1     = (const float*)d_in[19];
    const float* b_a1     = (const float*)d_in[20];
    const float* W_a2     = (const float*)d_in[21];
    const float* b_a2     = (const float*)d_in[22];
    const float* W_p1     = (const float*)d_in[23];
    const float* b_p1     = (const float*)d_in[24];
    const float* W_p2     = (const float*)d_in[25];
    const float* b_p2     = (const float*)d_in[26];
    const float* W_pol1   = (const float*)d_in[27];
    const float* b_pol1   = (const float*)d_in[28];
    const float* W_pol2   = (const float*)d_in[29];
    const float* b_pol2   = (const float*)d_in[30];
    const float* W_val1   = (const float*)d_in[31];
    const float* b_val1   = (const float*)d_in[32];
    const float* W_val2   = (const float*)d_in[33];
    const float* b_val2   = (const float*)d_in[34];
    const float* W_cf1    = (const float*)d_in[35];
    const float* b_cf1    = (const float*)d_in[36];
    const float* W_cf2    = (const float*)d_in[37];
    const float* b_cf2    = (const float*)d_in[38];

    float* out = (float*)d_out;
    char*  ws  = (char*)d_ws;
    _Float16* wm1_16 = (_Float16*)(ws + WS_WM1);
    _Float16* wih16  = (_Float16*)(ws + WS_WIH);
    _Float16* whh16  = (_Float16*)(ws + WS_WHH);
    _Float16* obj16  = (_Float16*)(ws + WS_OBJ);
    _Float16* x16    = (_Float16*)(ws + WS_X16);
    float* U      = (float*)(ws + WS_U);
    float* V      = (float*)(ws + WS_V);
    float* xg     = (float*)(ws + WS_XG);
    float* bcomb  = (float*)(ws + WS_BCOMB);
    int*   pres   = (int*)  (ws + WS_PRES);
    int*   chg    = (int*)  (ws + WS_CHG);
    float* cm_ws  = (float*)(ws + WS_CM);
    float* instr  = (float*)(ws + WS_INSTR);
    float* srep   = (float*)(ws + WS_SREP);
    float* crep   = (float*)(ws + WS_CREP);
    float* comb3  = (float*)(ws + WS_COMB3);
    float* comb   = (float*)(ws + WS_COMB);
    float* hpol   = (float*)(ws + WS_HPOL);
    float* hval   = (float*)(ws + WS_HVAL);
    float* hcf    = (float*)(ws + WS_HCF);
    float* hpar   = (float*)(ws + WS_HPAR);
    float* hal    = (float*)(ws + WS_HAL);
    float* scal   = (float*)(ws + WS_SCAL);
    float* hbuf   = (float*)(ws + WS_HBUF);
    int*   cnt    = (int*)  (ws + WS_CNT);

    const float* FNULL = nullptr;
    auto cdiv = [](int a, int b) { return (a + b - 1) / b; };

    // ---- 1) precision casts + gathers ----
    cvt_f32_f16<<<cdiv(2097152, 256), 256, 0, stream>>>(W_m1, wm1_16, 2097152);
    cvt_f32_f16<<<cdiv(4194304, 256), 256, 0, stream>>>(W_ih, wih16, 4194304);
    cvt_f32_f16<<<cdiv(4194304, 256), 256, 0, stream>>>(W_hh, whh16, 4194304);
    cvt_f32_f16<<<cdiv(65536, 256), 256, 0, stream>>>(obj_emb, obj16, 65536);
    gather_tokens<<<512, 256, 0, stream>>>(tokens, word_emb, x16);
    vec_add<<<16, 256, 0, stream>>>(b_ih, b_hh, bcomb, 4096);
    zero_int<<<1, 1, 0, stream>>>(cnt);

    // ---- 2) WMMA GEMMs ----
    // U = obj_emb @ W_m1[:, :1024]^T   (64x1024x1024)
    gemm_f16_wmma<<<32, 256, 0, stream>>>(obj16, 1024, wm1_16, 2048, U, 1024,
                                          64, 1024, 1024, FNULL, 0);
    // V = obj_emb @ W_m1[:, 1024:]^T
    gemm_f16_wmma<<<32, 256, 0, stream>>>(obj16, 1024, wm1_16 + 1024, 2048, V, 1024,
                                          64, 1024, 1024, FNULL, 0);
    // xg = x @ W_ih^T + (b_ih+b_hh)    (128x4096x1024)
    gemm_f16_wmma<<<256, 256, 0, stream>>>(x16, 1024, wih16, 1024, xg, 4096,
                                           128, 4096, 1024, bcomb, 0);

    // ---- 3) mechanism head -> causal_predictions (tiled 15x) ----
    mech_head<<<4032, 256, 0, stream>>>(U, V, b_m1, W_m2, b_m2, out + OUT_CP);

    // ---- 4) history -> causal matrix + matrix loss ----
    history_scan<<<15, 256, 0, stream>>>(hist, pres, chg);
    causal_mat_k<<<1, 1024, 0, stream>>>(causal_m, pres, chg, cm_ws, out + OUT_CM, scal);

    // ---- 5) bandwidth-bound GEMVs (f32) ----
    gemv_mb<<<128, 256, 0, stream>>>(state, 4096, W_enc, 4096, b_enc, srep, 1024,
                                     1, 1024, 4096, 0);
    gemv_mb<<<128, 256, 0, stream>>>(cm_ws, 4096, W_rand, 4096, FNULL, crep, 1024,
                                     1, 1024, 4096, 0);

    // ---- 6) multi-WGP sequential LSTM -> instr ----
    lstm_seq_mw<<<LSTM_NWG, 1024, 0, stream>>>(xg, whh16, hbuf, cnt, instr,
                                               out + OUT_INSTR);

    // ---- 7) language-side heads ----
    gemv_mb<<<128, 256, 0, stream>>>(instr, 1024, W_p1, 1024, b_p1, hpar, 1024,
                                     1, 1024, 1024, 1);
    gemv_mb<<<1, 256, 0, stream>>>(hpar, 1024, W_p2, 1024, b_p2, out + OUT_PARSE, 4,
                                   1, 4, 1024, 0);
    build_comb<<<64, 256, 0, stream>>>(instr, actions, act_emb, comb);
    gemv_mb<<<1024, 256, 0, stream>>>(comb, 2048, W_a1, 2048, b_a1, hal, 1024,
                                      8, 1024, 2048, 1);
    gemv_mb<<<1, 256, 0, stream>>>(hal, 1024, W_a2, 1024, b_a2, out + OUT_ALIGN, 1,
                                   8, 1, 1024, 0);
    grounding_k<<<1, 1024, 0, stream>>>(instr, actions, act_emb, out + OUT_GL, scal);

    // ---- 8) combined heads ----
    build_combined<<<12, 256, 0, stream>>>(srep, crep, instr, comb3);
    gemv_mb<<<128, 256, 0, stream>>>(comb3, 3072, W_pol1, 3072, b_pol1, hpol, 1024,
                                     1, 1024, 3072, 1);
    gemv_mb<<<1, 256, 0, stream>>>(hpol, 1024, W_pol2, 1024, b_pol2, out + OUT_POL, 8,
                                   1, 8, 1024, 0);
    gemv_mb<<<128, 256, 0, stream>>>(comb3, 3072, W_val1, 3072, b_val1, hval, 1024,
                                     1, 1024, 3072, 1);
    gemv_mb<<<1, 256, 0, stream>>>(hval, 1024, W_val2, 1024, b_val2, out + OUT_VAL, 1,
                                   1, 1, 1024, 0);
    gemv_mb<<<128, 256, 0, stream>>>(comb3, 3072, W_cf1, 3072, b_cf1, hcf, 1024,
                                     1, 1024, 3072, 1);
    gemv_mb<<<1, 256, 0, stream>>>(hcf, 1024, W_cf2, 1024, b_cf2, out + OUT_CF, 8,
                                   1, 8, 1024, 0);

    // ---- 9) total loss ----
    final_loss<<<1, 1, 0, stream>>>(scal, out + OUT_TL);
}
